// VisionTransformer_26285199852098
// MI455X (gfx1250) — compile-verified, matrix operand order checked by reading, not run on hardware
//
#include <hip/hip_runtime.h>
#include <math.h>

typedef __bf16 bf16;
typedef __attribute__((ext_vector_type(16))) __bf16 v16bf;
typedef __attribute__((ext_vector_type(8)))  float  v8f;
typedef __attribute__((ext_vector_type(4))) unsigned int u32x4;
typedef __attribute__((ext_vector_type(8))) int i32x8;
typedef __attribute__((ext_vector_type(4))) int i32x4;

#define WMMA_BF16(a,b,c) \
  __builtin_amdgcn_wmma_f32_16x16x32_bf16(false,(a),false,(b),(short)0,(c),false,false)

#if __has_builtin(__builtin_amdgcn_tensor_load_to_lds) && \
    __has_builtin(__builtin_amdgcn_s_wait_tensorcnt)
#define USE_TDM 1
#else
#define USE_TDM 0
#endif

// ---------------- model dims ----------------
#define DB    64
#define DD    768
#define DH    12
#define DL    12
#define DN    197
#define DHID  3072
#define MTOK  (DB*DN)    /* 12608 */
#define MPAT  (DB*196)   /* 12544 */
#define NPAD  224        /* keys padded to 14*16 */

__device__ __forceinline__ v8f vzero8() {
  v8f z;
#pragma unroll
  for (int i = 0; i < 8; ++i) z[i] = 0.f;
  return z;
}

// Load one 16x32 bf16 fragment row-slice for this lane.
// elements 0..7  = rowbase[khalf .. khalf+7]
// elements 8..15 = rowbase[khalf+16 .. khalf+23]
// (khalf = 0 for lanes 0-15, 8 for lanes 16-31) -> matches CDNA5 A/B layout.
__device__ __forceinline__ v16bf ld_frag(const bf16* rowbase, int khalf) {
  v16bf f;
  uint4* fp = reinterpret_cast<uint4*>(&f);
  fp[0] = *reinterpret_cast<const uint4*>(rowbase + khalf);
  fp[1] = *reinterpret_cast<const uint4*>(rowbase + khalf + 16);
  return f;
}

#if USE_TDM
// 2-D TDM tile load: global (strided rows) -> LDS (contiguous rows, optional
// 16B pad after every 64B for bank-conflict-free WMMA fragment loads).
// Units are 8-byte elements (data_size=3). Rows >= t1_rows are hardware
// zero-filled (TDM OOB semantics).  D# layout per CDNA5 ISA ch.8.
// NOTE: this toolchain exposes the 6-arg builtin form:
//   (u32x4 g0, i32x8 g1, i32x4 g2, i32x4 g3, i32x8 extra, i32 cpol)
__device__ __forceinline__ void tdm_load_2d(
    unsigned lds_addr, const void* gaddr,
    unsigned t0_units, unsigned t1_rows,
    unsigned tile0_units, unsigned tile1_rows,
    unsigned stride_units, int pad64_16)
{
  unsigned long long ga = (unsigned long long)gaddr;
  u32x4 g0;
  g0[0] = 1u;                                     // count=1 (valid user D#)
  g0[1] = lds_addr;                               // LDS byte address
  g0[2] = (unsigned)ga;                           // global_addr[31:0]
  g0[3] = (unsigned)(ga >> 32) | (2u << 30);      // global_addr[56:32] | type=2
  i32x8 g1;
  unsigned f0 = (3u << 16);                       // data_size = 8 bytes
  if (pad64_16) f0 |= (1u << 20) | (3u << 22) | (3u << 25); // pad: 64B interval, 16B amount
  g1[0] = (int)f0;
  g1[1] = (int)((t0_units & 0xFFFFu) << 16);                       // tensor_dim0[15:0]
  g1[2] = (int)(((t0_units >> 16) & 0xFFFFu) |
                ((t1_rows & 0xFFFFu) << 16));                      // dim0 hi | dim1 lo
  g1[3] = (int)(((t1_rows >> 16) & 0xFFFFu) | (tile0_units << 16)); // dim1 hi | tile_dim0
  g1[4] = (int)(tile1_rows & 0xFFFFu);                             // tile_dim1 (tile_dim2=0)
  g1[5] = (int)stride_units;                                       // tensor_dim0_stride lo
  g1[6] = 0;
  g1[7] = 0;
  i32x4 z4; z4[0] = 0; z4[1] = 0; z4[2] = 0; z4[3] = 0;
  i32x8 z8;
#pragma unroll
  for (int i = 0; i < 8; ++i) z8[i] = 0;
  __builtin_amdgcn_tensor_load_to_lds(g0, g1, z4, z4, z8, 0);
}
#endif

// ---------------- generic bf16 GEMM: C[M,N] = A[M,K] * W[N,K]^T (+bias,act,res) ----------------
// act: 0=none, 1=sin, 2=gelu(exact). Optional residual: C += res_scale*res[m,n] after act.
// Writes f32 (Cf) and/or bf16 (Cb).
__global__ __launch_bounds__(256) void k_gemm(
    const bf16* __restrict__ A, const bf16* __restrict__ W,
    const float* __restrict__ bias, const float* __restrict__ res, float res_scale,
    float* __restrict__ Cf, bf16* __restrict__ Cb,
    int M, int N, int K, int act)
{
#if USE_TDM
  __shared__ bf16 SMEM[2 * 2 * 128 * 40];        // [stage][A|W][128*40]
#else
  __shared__ bf16 SMEM[2 * 128 * 40];            // [A|W][128*40]
#endif
  const int tid  = threadIdx.x;
  const int lane = tid & 31, wave = tid >> 5;
  const int wm = wave >> 1, wn = wave & 1;       // 4x2 wave grid, 32x64 per wave
  const int gm0 = blockIdx.y * 128, gn0 = blockIdx.x * 128;
  const int lrow  = lane & 15;
  const int khalf = (lane >> 4) << 3;

  v8f acc[2][4];
#pragma unroll
  for (int mf = 0; mf < 2; ++mf)
#pragma unroll
    for (int nf = 0; nf < 4; ++nf) acc[mf][nf] = vzero8();

#if USE_TDM
  const unsigned smem_base = (unsigned)(size_t)SMEM;   // LDS byte offset of buffer
  const unsigned stride_units = (unsigned)K >> 2;      // K*2 bytes / 8
  const int nsteps = K >> 5;
  const unsigned rowsA = (unsigned)(M - gm0);          // remaining rows (OOB -> 0)
  const unsigned rowsW = (unsigned)(N - gn0);
  // issue stage 0
  if (wave == 0) {
    tdm_load_2d(smem_base,             A + (size_t)gm0 * K, (unsigned)K >> 2, rowsA,
                8, 128, stride_units, 1);
    tdm_load_2d(smem_base + 10240,     W + (size_t)gn0 * K, (unsigned)K >> 2, rowsW,
                8, 128, stride_units, 1);
  }
  for (int s = 0; s < nsteps; ++s) {
    const int stage = s & 1;
    if (wave == 0) {
      if (s + 1 < nsteps) {
        const int k1 = (s + 1) << 5;
        const unsigned rem = (unsigned)(K - k1) >> 2;
        tdm_load_2d(smem_base + (stage ^ 1) * 20480,
                    A + (size_t)gm0 * K + k1, rem, rowsA, 8, 128, stride_units, 1);
        tdm_load_2d(smem_base + (stage ^ 1) * 20480 + 10240,
                    W + (size_t)gn0 * K + k1, rem, rowsW, 8, 128, stride_units, 1);
        __builtin_amdgcn_s_wait_tensorcnt(2);   // current stage's pair complete
      } else {
        __builtin_amdgcn_s_wait_tensorcnt(0);
      }
    }
    __syncthreads();
    const bf16* As = SMEM + stage * 10240;
    const bf16* Ws = As + 5120;
    v16bf af[2], wf[4];
#pragma unroll
    for (int mf = 0; mf < 2; ++mf)
      af[mf] = ld_frag(&As[(wm * 32 + mf * 16 + lrow) * 40], khalf);
#pragma unroll
    for (int nf = 0; nf < 4; ++nf)
      wf[nf] = ld_frag(&Ws[(wn * 64 + nf * 16 + lrow) * 40], khalf);
#pragma unroll
    for (int mf = 0; mf < 2; ++mf)
#pragma unroll
      for (int nf = 0; nf < 4; ++nf)
        acc[mf][nf] = WMMA_BF16(af[mf], wf[nf], acc[mf][nf]);
    __syncthreads();
  }
#else
  bf16* As = SMEM;
  bf16* Ws = SMEM + 5120;
  for (int k0 = 0; k0 < K; k0 += 32) {
#pragma unroll
    for (int i = 0; i < 2; ++i) {
      int ch  = (tid << 1) | i;                  // 0..511
      int row = ch >> 2;
      int kc  = (ch & 3) << 3;                   // 0,8,16,24 halves
      int gr = gm0 + row; if (gr >= M) gr = M - 1;   // clamp: pad rows masked at store
      const bf16* ap = A + (size_t)gr * K + k0 + kc;
      *reinterpret_cast<uint4*>(&As[row * 40 + kc]) = *reinterpret_cast<const uint4*>(ap);
      if (k0 + 32 < K) __builtin_prefetch(ap + 32, 0, 1);
      int gn = gn0 + row; if (gn >= N) gn = N - 1;
      const bf16* wp = W + (size_t)gn * K + k0 + kc;
      *reinterpret_cast<uint4*>(&Ws[row * 40 + kc]) = *reinterpret_cast<const uint4*>(wp);
    }
    __syncthreads();
    v16bf af[2], wf[4];
#pragma unroll
    for (int mf = 0; mf < 2; ++mf)
      af[mf] = ld_frag(&As[(wm * 32 + mf * 16 + lrow) * 40], khalf);
#pragma unroll
    for (int nf = 0; nf < 4; ++nf)
      wf[nf] = ld_frag(&Ws[(wn * 64 + nf * 16 + lrow) * 40], khalf);
#pragma unroll
    for (int mf = 0; mf < 2; ++mf)
#pragma unroll
      for (int nf = 0; nf < 4; ++nf)
        acc[mf][nf] = WMMA_BF16(af[mf], wf[nf], acc[mf][nf]);
    __syncthreads();
  }
#endif

  const int rhi = (lane >> 4) << 3;              // +8 rows for lanes 16-31
#pragma unroll
  for (int mf = 0; mf < 2; ++mf) {
#pragma unroll
    for (int nf = 0; nf < 4; ++nf) {
      int col = gn0 + wn * 64 + nf * 16 + lrow;
      float bv = bias ? bias[col] : 0.f;
#pragma unroll
      for (int r = 0; r < 8; ++r) {
        int row = gm0 + wm * 32 + mf * 16 + rhi + r;
        if (row < M) {
          float v = acc[mf][nf][r] + bv;
          if (act == 1)       v = sinf(v);
          else if (act == 2)  v = 0.5f * v * (1.f + erff(v * 0.70710678118654752f));
          if (res) v += res_scale * res[(size_t)row * N + col];
          if (Cf) Cf[(size_t)row * N + col] = v;
          if (Cb) Cb[(size_t)row * N + col] = (bf16)v;
        }
      }
    }
  }
}

// ---------------- f32 -> bf16 (with scale, used for weights incl. EQ pre-scale) ----------------
__global__ __launch_bounds__(256) void k_cvt(const float* __restrict__ in,
                                             bf16* __restrict__ out, size_t n, float scale)
{
  size_t i = (size_t)blockIdx.x * 256 + threadIdx.x;
  if (i < n) out[i] = (bf16)(in[i] * scale);
}

// ---------------- im2col for 16x16 stride-16 conv ----------------
__global__ __launch_bounds__(256) void k_im2col(const float* __restrict__ x, bf16* __restrict__ Aout)
{
  size_t idx = (size_t)blockIdx.x * 256 + threadIdx.x;
  if (idx >= (size_t)MPAT * DD) return;
  int col = (int)(idx % DD);
  int row = (int)(idx / DD);
  int b = row / 196, pp = row % 196;
  int py = pp / 14, px = pp % 14;
  int c = col >> 8, ij = col & 255, ii = ij >> 4, jj = ij & 15;
  float v = x[(((size_t)b * 3 + c) * 224 + (py * 16 + ii)) * 224 + (px * 16 + jj)];
  Aout[idx] = (bf16)v;
}

// ---------------- assemble tokens: concat(cls, sin-patches) + pos ----------------
__global__ __launch_bounds__(256) void k_assemble(const float* __restrict__ patches,
                                                  const float* __restrict__ cls,
                                                  const float* __restrict__ pos,
                                                  float* __restrict__ tok)
{
  size_t idx = (size_t)blockIdx.x * 256 + threadIdx.x;
  if (idx >= (size_t)MTOK * DD) return;
  int d = (int)(idx % DD);
  int rest = (int)(idx / DD);
  int n = rest % DN, b = rest / DN;
  float v = (n == 0) ? cls[d] : patches[((size_t)b * 196 + (n - 1)) * DD + d];
  tok[idx] = v + pos[(size_t)n * DD + d];
}

// ---------------- self-modulated layernorm (param-free LN + per-sample affine) ----------------
__global__ __launch_bounds__(256) void k_smln(const float* __restrict__ T,
                                              const float* __restrict__ gam,
                                              const float* __restrict__ bet,
                                              float* __restrict__ outF, bf16* __restrict__ outB)
{
  const int row = blockIdx.x;            // 0..MTOK-1
  const int b = row / DN;
  const int tid = threadIdx.x;
  const float* xr = T + (size_t)row * DD;
  float v0 = xr[tid], v1 = xr[tid + 256], v2 = xr[tid + 512];
  float s = v0 + v1 + v2;
  float ss = v0 * v0 + v1 * v1 + v2 * v2;
#pragma unroll
  for (int m = 16; m >= 1; m >>= 1) {
    s  += __shfl_xor(s,  m, 32);
    ss += __shfl_xor(ss, m, 32);
  }
  __shared__ float shs[8], shss[8], stat[2];
  if ((tid & 31) == 0) { shs[tid >> 5] = s; shss[tid >> 5] = ss; }
  __syncthreads();
  if (tid == 0) {
    float S = 0.f, SS = 0.f;
    for (int w = 0; w < 8; ++w) { S += shs[w]; SS += shss[w]; }
    float mean = S * (1.f / 768.f);
    float var  = SS * (1.f / 768.f) - mean * mean;
    stat[0] = mean; stat[1] = rsqrtf(var + 1e-3f);
  }
  __syncthreads();
  float mean = stat[0], rs = stat[1];
  const float* gp = gam + (size_t)b * DD;
  const float* bp = bet + (size_t)b * DD;
#pragma unroll
  for (int i = 0; i < 3; ++i) {
    int d = tid + i * 256;
    float v = (i == 0) ? v0 : ((i == 1) ? v1 : v2);
    float y = (v - mean) * rs;
    y = y * (1.f + gp[d]) + bp[d];
    if (outF) outF[(size_t)row * DD + d] = y;
    if (outB) outB[(size_t)row * DD + d] = (bf16)y;
  }
}

// ---------------- fused attention: one WG per (b,h); WMMA scores + softmax + WMMA P*V ----------------
__global__ __launch_bounds__(256) void k_attn(const bf16* __restrict__ qkv, bf16* __restrict__ outb)
{
  const int bh = blockIdx.x;
  const int b = bh / DH, h = bh % DH;
  extern __shared__ char smem[];
  bf16* Kb = (bf16*)smem;            // [NPAD][64]   K rows (keys) x dims
  bf16* Vt = Kb + NPAD * 64;         // [64][NPAD]   V transposed: dims x keys
  bf16* Pb = Vt + 64 * NPAD;         // [8 waves][16][NPAD] prob tiles
  const int tid = threadIdx.x, wave = tid >> 5, lane = tid & 31;
  const int lrow = lane & 15, khalf = (lane >> 4) << 3;
  const size_t RS = 3 * DD;          // qkv row stride = 2304 elems

#if USE_TDM
  // K tile via Tensor Data Mover: 197 valid rows of 128B, pad rows zero-filled.
  if (wave == 0) {
    tdm_load_2d((unsigned)(size_t)Kb,
                qkv + (size_t)(b * DN) * RS + DD + h * 64,
                /*t0*/16, /*t1*/DN, /*tile0*/16, /*tile1*/NPAD,
                /*stride*/(unsigned)(RS * 2 / 8), /*pad*/0);
    __builtin_amdgcn_s_wait_tensorcnt(0);
  }
#else
  for (int ch = tid; ch < NPAD * 8; ch += 256) {
    int row = ch >> 3, c = (ch & 7) << 3;
    uint4 v = make_uint4(0, 0, 0, 0);
    if (row < DN)
      v = *reinterpret_cast<const uint4*>(qkv + (size_t)(b * DN + row) * RS + DD + h * 64 + c);
    *reinterpret_cast<uint4*>(&Kb[row * 64 + c]) = v;
  }
#endif
  // stage V transposed (scalar scatter)
  for (int e = tid; e < 64 * NPAD; e += 256) {
    int d = e / NPAD, n = e % NPAD;
    float v = 0.f;
    if (n < DN) v = (float)qkv[(size_t)(b * DN + n) * RS + 2 * DD + h * 64 + d];
    Vt[d * NPAD + n] = (bf16)v;
  }
  __syncthreads();

  bf16* Pw = Pb + wave * 16 * NPAD;
  const int rhi = (lane >> 4) << 3;

  for (int qt = wave; qt < 13; qt += 8) {            // 13 q-tiles of 16 (197 rows)
    int qrow = qt * 16 + lrow;
    int grow = b * DN + (qrow < DN ? qrow : DN - 1);
    const bf16* qp = qkv + (size_t)grow * RS + h * 64;
    v16bf aq0 = ld_frag(qp, khalf);                  // dims 0..31
    v16bf aq1 = ld_frag(qp + 32, khalf);             // dims 32..63

    v8f sc[14];
#pragma unroll
    for (int jt = 0; jt < 14; ++jt) {
      v8f s = vzero8();
      v16bf kf0 = ld_frag(&Kb[(jt * 16 + lrow) * 64], khalf);
      v16bf kf1 = ld_frag(&Kb[(jt * 16 + lrow) * 64 + 32], khalf);
      s = WMMA_BF16(aq0, kf0, s);
      s = WMMA_BF16(aq1, kf1, s);
      sc[jt] = s;
    }

    float mrow[8], srow[8];
#pragma unroll
    for (int r = 0; r < 8; ++r) mrow[r] = -3.0e38f;
#pragma unroll
    for (int jt = 0; jt < 14; ++jt) {
      bool ok = (jt * 16 + lrow) < DN;
#pragma unroll
      for (int r = 0; r < 8; ++r) {
        float v = ok ? sc[jt][r] * 0.125f : -3.0e38f;   // HD^-0.5 = 1/8
        sc[jt][r] = v;
        mrow[r] = fmaxf(mrow[r], v);
      }
    }
#pragma unroll
    for (int r = 0; r < 8; ++r) {
#pragma unroll
      for (int m = 1; m < 16; m <<= 1) mrow[r] = fmaxf(mrow[r], __shfl_xor(mrow[r], m, 32));
      srow[r] = 0.f;
    }
#pragma unroll
    for (int jt = 0; jt < 14; ++jt)
#pragma unroll
      for (int r = 0; r < 8; ++r) {
        float p = expf(sc[jt][r] - mrow[r]);
        sc[jt][r] = p;
        srow[r] += p;
      }
#pragma unroll
    for (int r = 0; r < 8; ++r) {
#pragma unroll
      for (int m = 1; m < 16; m <<= 1) srow[r] += __shfl_xor(srow[r], m, 32);
      srow[r] = 1.f / srow[r];
    }
    // C-layout -> A-layout bounce via LDS
#pragma unroll
    for (int jt = 0; jt < 14; ++jt)
#pragma unroll
      for (int r = 0; r < 8; ++r)
        Pw[(r + rhi) * NPAD + jt * 16 + lrow] = (bf16)(sc[jt][r] * srow[r]);

    v8f o[4];
#pragma unroll
    for (int df = 0; df < 4; ++df) o[df] = vzero8();
#pragma unroll
    for (int kc = 0; kc < 7; ++kc) {                 // 224 keys / 32
      v16bf pf = ld_frag(&Pw[lrow * NPAD + kc * 32], khalf);
#pragma unroll
      for (int df = 0; df < 4; ++df) {
        v16bf vf = ld_frag(&Vt[(df * 16 + lrow) * NPAD + kc * 32], khalf);
        o[df] = WMMA_BF16(pf, vf, o[df]);
      }
    }
#pragma unroll
    for (int df = 0; df < 4; ++df) {
      int col = h * 64 + df * 16 + lrow;
#pragma unroll
      for (int r = 0; r < 8; ++r) {
        int qr = qt * 16 + rhi + r;
        if (qr < DN)
          outb[(size_t)(b * DN + qr) * DD + col] = (bf16)o[df][r];
      }
    }
  }
}

// ---------------- host orchestration ----------------
extern "C" void kernel_launch(void* const* d_in, const int* in_sizes, int n_in,
                              void* d_out, int out_size, void* d_ws, size_t ws_size,
                              hipStream_t stream)
{
  (void)in_sizes; (void)n_in; (void)out_size; (void)ws_size;
  const float* x      = (const float*)d_in[0];
  const float* latent = (const float*)d_in[1];
  const float* conv_w = (const float*)d_in[2];
  const float* conv_b = (const float*)d_in[3];
  const float* cls    = (const float*)d_in[4];
  const float* pos    = (const float*)d_in[5];
  const float* g1w = (const float*)d_in[6];  const float* g1b = (const float*)d_in[7];
  const float* b1w = (const float*)d_in[8];  const float* b1b = (const float*)d_in[9];
  const float* qkvw = (const float*)d_in[10];
  const float* projw = (const float*)d_in[11]; const float* projb = (const float*)d_in[12];
  const float* g2w = (const float*)d_in[13]; const float* g2b = (const float*)d_in[14];
  const float* b2w = (const float*)d_in[15]; const float* b2b = (const float*)d_in[16];
  const float* fc1w = (const float*)d_in[17]; const float* fc1b = (const float*)d_in[18];
  const float* fc2w = (const float*)d_in[19]; const float* fc2b = (const float*)d_in[20];
  const float* gnw = (const float*)d_in[21]; const float* gnb = (const float*)d_in[22];
  const float* bnw = (const float*)d_in[23]; const float* bnb = (const float*)d_in[24];
  float* OUT = (float*)d_out;

  // workspace carve
  char* p = (char*)d_ws;
  auto alloc = [&](size_t bytes) -> char* {
    char* r = p; p += (bytes + 255) & ~(size_t)255; return r;
  };
  float* T   = (float*)alloc((size_t)MTOK * DD * 4);     // token stream (f32)
  float* XN  = (float*)alloc((size_t)MTOK * DD * 4);     // smln output (f32, residual src)
  bf16*  XNb = (bf16*) alloc((size_t)MTOK * DD * 2);     // smln output (bf16, GEMM A)
  bf16*  ACT = (bf16*) alloc((size_t)MTOK * DD * 2);     // im2col A / attention out (bf16)
  char*  BIG =         alloc((size_t)MTOK * DHID * 2);   // patches f32 / qkv bf16 / gelu bf16
  bf16*  WBUF= (bf16*) alloc((size_t)DHID * DD * 2);     // weight bf16 scratch
  bf16*  LATB= (bf16*) alloc((size_t)DB * DD * 2);
  float* GAM = (float*)alloc((size_t)DB * DD * 4);
  float* BET = (float*)alloc((size_t)DB * DD * 4);

  const float EQ = 0.036084391824351615f;                // 1/sqrt(768)
  auto cvt = [&](const float* in, bf16* out, size_t n, float sc) {
    k_cvt<<<dim3((unsigned)((n + 255) / 256)), 256, 0, stream>>>(in, out, n, sc);
  };
  auto gemm = [&](const bf16* A, const bf16* W, const float* bias, const float* res,
                  float rsc, float* Cf, bf16* Cb, int M, int N, int K, int act) {
    dim3 g(N / 128, (M + 127) / 128);
    k_gemm<<<g, 256, 0, stream>>>(A, W, bias, res, rsc, Cf, Cb, M, N, K, act);
  };
  auto smln = [&](const float* Ti, const float* ga, const float* be, float* oF, bf16* oB) {
    k_smln<<<dim3(MTOK), 256, 0, stream>>>(Ti, ga, be, oF, oB);
  };

  const size_t DxD = (size_t)DD * DD;

  // latent -> bf16
  cvt(latent, LATB, (size_t)DB * DD, 1.f);

  // patch embed: im2col -> GEMM(sin) -> assemble tokens
  k_im2col<<<dim3((unsigned)(((size_t)MPAT * DD + 255) / 256)), 256, 0, stream>>>(x, ACT);
  cvt(conv_w, WBUF, DxD, 1.f);
  gemm(ACT, WBUF, conv_b, nullptr, 0.f, (float*)BIG, nullptr, MPAT, DD, DD, /*sin*/1);
  k_assemble<<<dim3((unsigned)(((size_t)MTOK * DD + 255) / 256)), 256, 0, stream>>>(
      (const float*)BIG, cls, pos, T);

  for (int l = 0; l < DL; ++l) {
    // modulation 1
    cvt(g1w + (size_t)l * DxD, WBUF, DxD, EQ);
    gemm(LATB, WBUF, g1b + (size_t)l * DD, nullptr, 0.f, GAM, nullptr, DB, DD, DD, 0);
    cvt(b1w + (size_t)l * DxD, WBUF, DxD, EQ);
    gemm(LATB, WBUF, b1b + (size_t)l * DD, nullptr, 0.f, BET, nullptr, DB, DD, DD, 0);
    smln(T, GAM, BET, XN, XNb);                       // xn (f32 + bf16)

    // qkv -> bf16 straight into BIG
    cvt(qkvw + (size_t)l * 3 * DxD, WBUF, 3 * DxD, 1.f);
    gemm(XNb, WBUF, nullptr, nullptr, 0.f, nullptr, (bf16*)BIG, MTOK, 3 * DD, DD, 0);

    // attention (K via TDM, V/P in LDS, WMMA both matmuls)
    k_attn<<<dim3(DB * DH), 256, (NPAD * 64 + 64 * NPAD + 8 * 16 * NPAD) * 2, stream>>>(
        (const bf16*)BIG, ACT);

    // proj with fused bias + 2*xn residual (source quirk: t = 2*xn + attn_out)
    cvt(projw + (size_t)l * DxD, WBUF, DxD, 1.f);
    gemm(ACT, WBUF, projb + (size_t)l * DD, XN, 2.f, T, nullptr, MTOK, DD, DD, 0);

    // modulation 2
    cvt(g2w + (size_t)l * DxD, WBUF, DxD, EQ);
    gemm(LATB, WBUF, g2b + (size_t)l * DD, nullptr, 0.f, GAM, nullptr, DB, DD, DD, 0);
    cvt(b2w + (size_t)l * DxD, WBUF, DxD, EQ);
    gemm(LATB, WBUF, b2b + (size_t)l * DD, nullptr, 0.f, BET, nullptr, DB, DD, DD, 0);
    smln(T, GAM, BET, XN, XNb);                       // t2 (f32 + bf16)

    // MLP: fc1 (gelu, bf16 out into BIG), fc2 (+bias, +t2 residual)
    cvt(fc1w + (size_t)l * DD * DHID, WBUF, (size_t)DD * DHID, 1.f);
    gemm(XNb, WBUF, fc1b + (size_t)l * DHID, nullptr, 0.f, nullptr, (bf16*)BIG,
         MTOK, DHID, DD, /*gelu*/2);
    cvt(fc2w + (size_t)l * DD * DHID, WBUF, (size_t)DD * DHID, 1.f);
    gemm((const bf16*)BIG, WBUF, fc2b + (size_t)l * DD, XN, 1.f, T, nullptr,
         MTOK, DD, DHID, 0);
  }

  // final smln -> d_out
  cvt(gnw, WBUF, DxD, EQ);
  gemm(LATB, WBUF, gnb, nullptr, 0.f, GAM, nullptr, DB, DD, DD, 0);
  cvt(bnw, WBUF, DxD, EQ);
  gemm(LATB, WBUF, bnb, nullptr, 0.f, BET, nullptr, DB, DD, DD, 0);
  smln(T, GAM, BET, OUT, nullptr);
}
